// GCN_10934986735728
// MI455X (gfx1250) — matrix operand order repaired and checked by reading, not compiled, not run on hardware
//
#include <hip/hip_runtime.h>
#include <math.h>

typedef __attribute__((ext_vector_type(16))) __bf16 bf16x16;
typedef __attribute__((ext_vector_type(8)))  float  floatx8;

#define WG 256
#define BN_EPS 1e-5f

// gfx1250 async global->LDS copy (CDNA5 ISA): per-lane 16B
// DMA into LDS, tracked by ASYNCcnt, no VGPR round-trip.
#define ASYNC_LOAD_B128(lds_u32, gaddr_u64)                                    \
  asm volatile("global_load_async_to_lds_b128 %0, %1, off"                     \
               :: "v"(lds_u32), "v"(gaddr_u64) : "memory")
#define WAIT_ASYNCCNT0() asm volatile("s_wait_asynccnt 0" ::: "memory")

// ---------------------------------------------------------------- utilities
__global__ void k_zero(float* __restrict__ p, long n) {
  long i = (long)blockIdx.x * blockDim.x + threadIdx.x;
  long s = (long)gridDim.x * blockDim.x;
  for (; i < n; i += s) p[i] = 0.0f;
}

__global__ void k_deg(const int* __restrict__ src, float* __restrict__ deg, int E) {
  int i = blockIdx.x * blockDim.x + threadIdx.x;
  if (i < E)
    __hip_atomic_fetch_add(&deg[src[i]], 1.0f, __ATOMIC_RELAXED, __HIP_MEMORY_SCOPE_AGENT);
}

__global__ void k_dis(float* __restrict__ deg, int N) {
  int i = blockIdx.x * blockDim.x + threadIdx.x;
  if (i < N) { float d = deg[i]; deg[i] = (d > 0.0f) ? rsqrtf(d) : 0.0f; }
}

__global__ void k_ew(const int* __restrict__ src, const int* __restrict__ dst,
                     const float* __restrict__ dis, float* __restrict__ ew, int E) {
  int i = blockIdx.x * blockDim.x + threadIdx.x;
  if (i < E) ew[i] = -dis[src[i]] * dis[dst[i]];
}

// ------------------------------------------------- edge propagation (scatter)
// Block = (WG>>lgG) edges x (1<<lgG) float4 chunks; consecutive threads cover
// consecutive chunks of one edge -> fully coalesced 256B gathers, no division.
__global__ void k_prop(const float* __restrict__ h, const int* __restrict__ src,
                       const int* __restrict__ dst, const float* __restrict__ ew,
                       float* __restrict__ out, int E, int C, int lgG) {
  const int groups = C >> 2;
  const int c4 = (threadIdx.x & (groups - 1)) << 2;
  const int epb = WG >> lgG;
  const int estep = gridDim.x * epb;
  int e = blockIdx.x * epb + (threadIdx.x >> lgG);
  for (; e < E; e += estep) {
    float w = ew[e];
    const float4 v = *reinterpret_cast<const float4*>(&h[(size_t)src[e] * C + c4]);
    float* o = &out[(size_t)dst[e] * C + c4];
    __hip_atomic_fetch_add(&o[0], w * v.x, __ATOMIC_RELAXED, __HIP_MEMORY_SCOPE_AGENT);
    __hip_atomic_fetch_add(&o[1], w * v.y, __ATOMIC_RELAXED, __HIP_MEMORY_SCOPE_AGENT);
    __hip_atomic_fetch_add(&o[2], w * v.z, __ATOMIC_RELAXED, __HIP_MEMORY_SCOPE_AGENT);
    __hip_atomic_fetch_add(&o[3], w * v.w, __ATOMIC_RELAXED, __HIP_MEMORY_SCOPE_AGENT);
  }
}

// ----------------------------------------------- weight pre-pack (fp32->bf16)
// Wp[(((kt*ntiles)+nt)*32+lane)*16+i] = W[(kt*32+(lane<16?0:8)+i+(i<8?0:8))*M + nt*16+(lane&15)]
__global__ void k_pack_w(const float* __restrict__ W, __bf16* __restrict__ Wp,
                         int K, int M) {
  int ntiles = M >> 4;
  int total = K * M;
  int t = blockIdx.x * blockDim.x + threadIdx.x;
  if (t >= total) return;
  int i    = t & 15;
  int lane = (t >> 4) & 31;
  int rest = t >> 9;
  int nt   = rest % ntiles;
  int kt   = rest / ntiles;
  int n = nt * 16 + (lane & 15);
  int k = kt * 32 + ((lane < 16) ? 0 : 8) + i + ((i < 8) ? 0 : 8);
  Wp[t] = (__bf16)W[(size_t)k * M + n];
}

// A fragment (16x32 bf16, ISA 7.12.2) built from an fp32 LDS row.
__device__ __forceinline__ bf16x16 make_afrag(const float* __restrict__ arow, int kbA) {
  float4 f0 = *(const float4*)&arow[kbA];
  float4 f1 = *(const float4*)&arow[kbA + 4];
  float4 f2 = *(const float4*)&arow[16 + kbA];
  float4 f3 = *(const float4*)&arow[16 + kbA + 4];
  bf16x16 a;
  a[0] = (__bf16)f0.x; a[1] = (__bf16)f0.y; a[2]  = (__bf16)f0.z; a[3]  = (__bf16)f0.w;
  a[4] = (__bf16)f1.x; a[5] = (__bf16)f1.y; a[6]  = (__bf16)f1.z; a[7]  = (__bf16)f1.w;
  a[8] = (__bf16)f2.x; a[9] = (__bf16)f2.y; a[10] = (__bf16)f2.z; a[11] = (__bf16)f2.w;
  a[12] = (__bf16)f3.x; a[13] = (__bf16)f3.y; a[14] = (__bf16)f3.z; a[15] = (__bf16)f3.w;
  return a;
}

// -------------------------------------------------------- WMMA bf16 GEMM
// C[N x M] (+)= A[N x K] @ W[K x M] + bias.
// Block of (nwaves) waves covers 32 rows x (nwaves*16) cols; every wave is
// active (no EXEC divergence around WMMA).  Each wave owns two 16x16 D tiles
// (independent back-to-back WMMAs sharing one B fragment).  The 32x32 fp32 A
// tile (4KB = 256 x 16B chunks) is staged by async global->LDS DMA.
__global__ void __launch_bounds__(WG)
k_gemm_wmma(const float* __restrict__ A, const __bf16* __restrict__ Wp,
            const float* __restrict__ bias, float* __restrict__ C,
            int K, int M, int accflag) {
  __shared__ float As32[32 * 32];
  const int lane = threadIdx.x & 31;
  const int wave = threadIdx.x >> 5;
  const int row0 = blockIdx.x << 5;
  const int col0 = blockIdx.y * (blockDim.x >> 1) + (wave << 4);
  const int ntiles = M >> 4;
  const int ntile  = col0 >> 4;

  floatx8 acc0 = {0.f,0.f,0.f,0.f,0.f,0.f,0.f,0.f};
  floatx8 acc1 = {0.f,0.f,0.f,0.f,0.f,0.f,0.f,0.f};

  const int mrow = lane & 15;
  const int kbA  = (lane < 16) ? 0 : 8;        // A fragment K-base

  const bf16x16* __restrict__ WpV = (const bf16x16*)Wp;

  for (int k0 = 0, kt = 0; k0 < K; k0 += 32, ++kt) {
    for (int t = threadIdx.x; t < 256; t += blockDim.x) {
      int sm = t >> 3;                          // row 0..31
      int sq = (t & 7) << 2;                    // col 0,4,...,28
      unsigned lds_dst = (unsigned)(size_t)&As32[sm * 32 + sq];
      unsigned long long ga =
          (unsigned long long)(size_t)&A[(size_t)(row0 + sm) * K + k0 + sq];
      ASYNC_LOAD_B128(lds_dst, ga);
    }
    if (k0 + 32 < K)
      __builtin_prefetch(&A[(size_t)row0 * K + k0 + 32], 0, 1);
    WAIT_ASYNCCNT0();
    __syncthreads();

    bf16x16 b  = WpV[((size_t)kt * ntiles + ntile) * 32 + lane];
    bf16x16 a0 = make_afrag(&As32[mrow * 32], kbA);
    bf16x16 a1 = make_afrag(&As32[(16 + mrow) * 32], kbA);
    acc0 = __builtin_amdgcn_wmma_f32_16x16x32_bf16(
        false, a0, false, b, (short)0, acc0, false, false);
    acc1 = __builtin_amdgcn_wmma_f32_16x16x32_bf16(
        false, a1, false, b, (short)0, acc1, false, false);

    __syncthreads();                            // protect As32 before next DMA
  }

  const int ncol = col0 + (lane & 15);
  const float bv = bias[ncol];                  // bias is always valid
  if (accflag) {
#pragma unroll
    for (int r = 0; r < 8; ++r) {
      int m = (lane < 16) ? r : (r + 8);        // C/D layout (ISA 7.12.2)
      C[(size_t)(row0 + m) * M + ncol]      += acc0[r] + bv;
      C[(size_t)(row0 + 16 + m) * M + ncol] += acc1[r] + bv;
    }
  } else {
#pragma unroll
    for (int r = 0; r < 8; ++r) {
      int m = (lane < 16) ? r : (r + 8);
      C[(size_t)(row0 + m) * M + ncol]      = acc0[r] + bv;
      C[(size_t)(row0 + 16 + m) * M + ncol] = acc1[r] + bv;
    }
  }
}

// ---------------------------------------------------------------- batch norm
__global__ void k_bn_partial(const float* __restrict__ h, float* __restrict__ sum,
                             float* __restrict__ sum2, int Nrows, int C) {
  const int c  = blockIdx.y * 32 + (threadIdx.x & 31);
  const int r0 = blockIdx.x * 8 + (threadIdx.x >> 5);
  const int rstep = 8 * gridDim.x;
  float s = 0.f, s2 = 0.f;
  for (int r = r0; r < Nrows; r += rstep) {
    float v = h[(size_t)r * C + c];
    s += v; s2 += v * v;
  }
  __shared__ float sh[WG], sh2[WG];
  sh[threadIdx.x] = s; sh2[threadIdx.x] = s2;
  __syncthreads();
  if (threadIdx.x < 32) {
    float a = 0.f, b = 0.f;
#pragma unroll
    for (int j = 0; j < 8; ++j) { a += sh[threadIdx.x + 32 * j]; b += sh2[threadIdx.x + 32 * j]; }
    __hip_atomic_fetch_add(&sum[c],  a, __ATOMIC_RELAXED, __HIP_MEMORY_SCOPE_AGENT);
    __hip_atomic_fetch_add(&sum2[c], b, __ATOMIC_RELAXED, __HIP_MEMORY_SCOPE_AGENT);
  }
}

// sums -> (shift, scale):  y = x*scale + shift  ==  (x-m)*rsqrt(v+eps)*g + be
// With g == nullptr, just writes the mean (used for the global mean pool).
__global__ void k_bn_finalize(float* __restrict__ sum, float* __restrict__ sum2,
                              const float* __restrict__ g, const float* __restrict__ be,
                              int Nrows, int C) {
  int c = blockIdx.x * blockDim.x + threadIdx.x;
  if (c < C) {
    float m = sum[c] / (float)Nrows;
    if (g) {
      float sc = rsqrtf(sum2[c] / (float)Nrows - m * m + BN_EPS) * g[c];
      sum2[c] = sc;                 // scale
      sum[c]  = be[c] - m * sc;     // shift
    } else {
      sum[c] = m;
    }
  }
}

__global__ void k_bn_apply(float* __restrict__ h, const float* __restrict__ shift,
                           const float* __restrict__ scale, long total4, int C, int relu) {
  const int g4 = C >> 2;
  long i = (long)blockIdx.x * blockDim.x + threadIdx.x;
  long stride = (long)gridDim.x * blockDim.x;
  float4* h4 = (float4*)h;
  for (; i < total4; i += stride) {
    int c4 = ((int)(i % g4)) << 2;
    float4 v = h4[i];
    float* vp = &v.x;
#pragma unroll
    for (int j = 0; j < 4; ++j) {
      float t = fmaf(vp[j], scale[c4 + j], shift[c4 + j]);
      vp[j] = relu ? fmaxf(t, 0.0f) : t;
    }
    h4[i] = v;
  }
}

// ------------------------------------------------------------------ MLP head
__global__ void k_head(const float* __restrict__ pooled,
                       const float* __restrict__ lw1, const float* __restrict__ lb1,
                       const float* __restrict__ lw2, const float* __restrict__ lb2,
                       const float* __restrict__ lw3, const float* __restrict__ lb3,
                       float* __restrict__ out) {
  __shared__ float g[512], t1[128], t2[64], t3[10];
  const int t = threadIdx.x;
  g[t] = pooled[t];
  __syncthreads();
  if (t < 128) { float s = lb1[t]; for (int k = 0; k < 512; ++k) s += g[k] * lw1[k * 128 + t]; t1[t] = s; }
  __syncthreads();
  if (t < 64)  { float s = lb2[t]; for (int k = 0; k < 128; ++k) s += t1[k] * lw2[k * 64 + t];  t2[t] = s; }
  __syncthreads();
  if (t < 10)  { float s = lb3[t]; for (int k = 0; k < 64;  ++k) s += t2[k] * lw3[k * 10 + t];  t3[t] = s; }
  __syncthreads();
  if (t == 0) {
    float mx = t3[0];
    for (int i = 1; i < 10; ++i) mx = fmaxf(mx, t3[i]);
    float se = 0.f;
    for (int i = 0; i < 10; ++i) se += expf(t3[i] - mx);
    float lse = mx + logf(se);
    for (int i = 0; i < 10; ++i) out[i] = t3[i] - lse;
  }
}

// ------------------------------------------------------------------ launcher
extern "C" void kernel_launch(void* const* d_in, const int* in_sizes, int n_in,
                              void* d_out, int out_size, void* d_ws, size_t ws_size,
                              hipStream_t stream) {
  const float* x    = (const float*)d_in[0];
  const int*   ei   = (const int*)  d_in[1];
  const float* w1_0 = (const float*)d_in[2];
  const float* b1   = (const float*)d_in[3];
  const float* w2_0 = (const float*)d_in[4];
  const float* w2_1 = (const float*)d_in[5];
  const float* b2   = (const float*)d_in[6];
  const float* w3_0 = (const float*)d_in[7];
  const float* w3_1 = (const float*)d_in[8];
  const float* b3   = (const float*)d_in[9];
  const float* g1   = (const float*)d_in[10];
  const float* be1  = (const float*)d_in[11];
  const float* g2   = (const float*)d_in[12];
  const float* be2  = (const float*)d_in[13];
  const float* g3   = (const float*)d_in[14];
  const float* be3  = (const float*)d_in[15];
  const float* lw1  = (const float*)d_in[16];
  const float* lb1  = (const float*)d_in[17];
  const float* lw2  = (const float*)d_in[18];
  const float* lb2  = (const float*)d_in[19];
  const float* lw3  = (const float*)d_in[20];
  const float* lb3  = (const float*)d_in[21];
  float* out = (float*)d_out;

  const int N = in_sizes[0] / 128;   // 100000
  const int E = in_sizes[1] / 2;     // 1600000
  const int* src = ei;
  const int* dst = ei + E;

  char* ws = (char*)d_ws;
  size_t off = 0;
  auto allocf = [&](size_t nflt) {
    float* p = (float*)(ws + off);
    off += ((nflt * sizeof(float) + 255) / 256) * 256;
    return p;
  };
  auto allocb = [&](size_t nbf) {
    __bf16* p = (__bf16*)(ws + off);
    off += ((nbf * sizeof(__bf16) + 255) / 256) * 256;
    return p;
  };
  float* h1    = allocf((size_t)N * 64);
  float* p1    = allocf((size_t)N * 64);
  float* h2    = allocf((size_t)N * 128);
  float* p2    = allocf((size_t)N * 128);
  float* h3    = allocf((size_t)N * 512);
  float* dis   = allocf((size_t)N);        // deg, then D^{-1/2}
  float* ew    = allocf((size_t)E);
  float* stats = allocf(1024);             // shift/mean | scale/var
  float* mean  = stats;
  float* var   = stats + 512;
  float* zbias = allocf(512);              // zero bias for accumulate GEMMs
  __bf16* Wp1  = allocb(128 * 64);
  __bf16* Wp20 = allocb(64 * 128);
  __bf16* Wp21 = allocb(64 * 128);
  __bf16* Wp30 = allocb(128 * 512);
  __bf16* Wp31 = allocb(128 * 512);
  (void)ws_size; (void)n_in; (void)out_size;

  const int ZB = 4096;

  // one-time weight pre-pack into bf16 fragment order + zero bias slab
  k_pack_w<<<(128 * 64  + WG - 1) / WG, WG, 0, stream>>>(w1_0, Wp1,  128, 64);
  k_pack_w<<<(64 * 128  + WG - 1) / WG, WG, 0, stream>>>(w2_0, Wp20, 64, 128);
  k_pack_w<<<(64 * 128  + WG - 1) / WG, WG, 0, stream>>>(w2_1, Wp21, 64, 128);
  k_pack_w<<<(128 * 512 + WG - 1) / WG, WG, 0, stream>>>(w3_0, Wp30, 128, 512);
  k_pack_w<<<(128 * 512 + WG - 1) / WG, WG, 0, stream>>>(w3_1, Wp31, 128, 512);
  k_zero<<<2, WG, 0, stream>>>(zbias, 512);

  // edge weights: -D^{-1/2} A D^{-1/2}
  k_zero<<<ZB, WG, 0, stream>>>(dis, (long)N);
  k_deg<<<(E + WG - 1) / WG, WG, 0, stream>>>(src, dis, E);
  k_dis<<<(N + WG - 1) / WG, WG, 0, stream>>>(dis, N);
  k_ew<<<(E + WG - 1) / WG, WG, 0, stream>>>(src, dst, dis, ew, E);

  const int MT = N / 32;                   // 3125 row tiles (exact)
  const int RB = 64;                       // row-blocks for BN partial pass

  // conv1 (K=1): h1 = x @ w1_0 + b1 ; bn1 ; relu   (M=64 -> 4-wave blocks)
  k_gemm_wmma<<<dim3(MT, 1), 128, 0, stream>>>(x, Wp1, b1, h1, 128, 64, 0);
  k_zero<<<4, WG, 0, stream>>>(stats, 1024);
  k_bn_partial<<<dim3(RB, 64 / 32), WG, 0, stream>>>(h1, mean, var, N, 64);
  k_bn_finalize<<<2, WG, 0, stream>>>(mean, var, g1, be1, N, 64);
  k_bn_apply<<<ZB, WG, 0, stream>>>(h1, mean, var, (long)N * 16, 64, 1);

  // conv2 (K=2): h2 = h1 @ w2_0 + prop(h1) @ w2_1 + b2 ; bn2 ; relu
  k_zero<<<ZB, WG, 0, stream>>>(p1, (long)N * 64);
  k_prop<<<8192, WG, 0, stream>>>(h1, src, dst, ew, p1, E, 64, 4);
  k_gemm_wmma<<<dim3(MT, 1), WG, 0, stream>>>(h1, Wp20, b2, h2, 64, 128, 0);
  k_gemm_wmma<<<dim3(MT, 1), WG, 0, stream>>>(p1, Wp21, zbias, h2, 64, 128, 1);
  k_zero<<<4, WG, 0, stream>>>(stats, 1024);
  k_bn_partial<<<dim3(RB, 128 / 32), WG, 0, stream>>>(h2, mean, var, N, 128);
  k_bn_finalize<<<2, WG, 0, stream>>>(mean, var, g2, be2, N, 128);
  k_bn_apply<<<ZB, WG, 0, stream>>>(h2, mean, var, (long)N * 32, 128, 1);

  // conv3 (K=2): h3 = h2 @ w3_0 + prop(h2) @ w3_1 + b3 ; bn3 (no relu)
  k_zero<<<ZB, WG, 0, stream>>>(p2, (long)N * 128);
  k_prop<<<8192, WG, 0, stream>>>(h2, src, dst, ew, p2, E, 128, 5);
  k_gemm_wmma<<<dim3(MT, 4), WG, 0, stream>>>(h2, Wp30, b3, h3, 128, 512, 0);
  k_gemm_wmma<<<dim3(MT, 4), WG, 0, stream>>>(p2, Wp31, zbias, h3, 128, 512, 1);
  k_zero<<<4, WG, 0, stream>>>(stats, 1024);
  k_bn_partial<<<dim3(RB, 512 / 32), WG, 0, stream>>>(h3, mean, var, N, 512);
  k_bn_finalize<<<2, WG, 0, stream>>>(mean, var, g3, be3, N, 512);
  k_bn_apply<<<ZB, WG, 0, stream>>>(h3, mean, var, (long)N * 128, 512, 0);

  // global mean pool == per-channel mean of normalized h3
  k_zero<<<4, WG, 0, stream>>>(stats, 1024);
  k_bn_partial<<<dim3(RB, 512 / 32), WG, 0, stream>>>(h3, mean, var, N, 512);
  k_bn_finalize<<<2, WG, 0, stream>>>(mean, var, nullptr, nullptr, N, 512);
  k_head<<<1, 512, 0, stream>>>(mean, lw1, lb1, lw2, lb2, lw3, lb3, out);
}
